// SegmentationHead_61881888801118
// MI455X (gfx1250) — matrix operand order repaired and checked by reading, not compiled, not run on hardware
//
#include <hip/hip_runtime.h>

#define DD    2048
#define ROWS  16
#define PADD  2056          // padded LDS row stride (elements) -> conflict-free ds_load_b128
#define NT    16            // 16x16 N-tiles per wave (wave owns 256 columns)
#define EPSF  1e-5f

typedef __attribute__((ext_vector_type(16))) __bf16 v16bf;
typedef __attribute__((ext_vector_type(8)))  __bf16 v8bf;
typedef __attribute__((ext_vector_type(4)))  __bf16 v4bf;
typedef __attribute__((ext_vector_type(8)))  float  v8f;

struct Params {
    const float*  x;
    const __bf16* W1[3];
    const float*  b1[3];
    const float*  g[3];
    const float*  be[3];
    const float*  W2[3];
    const float*  b2[3];
    float*        out;
};

// ---- prep: f32 -> bf16 conversion of W1 (stays resident in 192MB L2 afterwards) ----
__global__ __launch_bounds__(256) void cvt_f32_bf16(const float* __restrict__ in,
                                                    __bf16* __restrict__ out, int n4) {
    int i = blockIdx.x * 256 + threadIdx.x;
    if (i >= n4) return;
    float4 v = ((const float4*)in)[i];
    v4bf o;
    o[0] = (__bf16)v.x; o[1] = (__bf16)v.y; o[2] = (__bf16)v.z; o[3] = (__bf16)v.w;
    ((v4bf*)out)[i] = o;
}

// ---- fused: 3x (GEMM -> +b1 -> LN -> ReLU -> dot W2) -> route -> select ----
__global__ __launch_bounds__(256) void fused_moe_head(Params p) {
    __shared__ __bf16 xs[ROWS * PADD];          // 65792 B  x tile (bf16)
    __shared__ float  red[8 * 32 * 16];         // 16384 B  cross-wave reduction scratch
    __shared__ float  mu[ROWS], rs[ROWS];
    __shared__ float  logits[3][ROWS][2];

    const int tid  = threadIdx.x;
    const int wave = tid >> 5;
    const int lane = tid & 31;
    const int lo   = lane & 15;                 // M index (A) or N index (B/C)
    const int hi   = lane >> 4;                 // K-half / M-half selector
    const size_t row0 = (size_t)blockIdx.x * ROWS;
    const int colbase = wave * (NT * 16);       // this wave's N range

    // ---- stage X tile into LDS as bf16 ----
    {
        const float* xp = p.x + row0 * DD;
        for (int i = tid; i < ROWS * (DD / 4); i += 256) {
            int r  = i >> 9;                    // / (DD/4)
            int c4 = i & 511;
            float4 v = ((const float4*)xp)[(size_t)r * (DD / 4) + c4];
            v4bf o;
            o[0] = (__bf16)v.x; o[1] = (__bf16)v.y; o[2] = (__bf16)v.z; o[3] = (__bf16)v.w;
            *(v4bf*)(xs + r * PADD + c4 * 4) = o;
        }
    }
    __syncthreads();

#pragma unroll
    for (int br = 0; br < 3; ++br) {
        const __bf16* __restrict__ Wb = p.W1[br];

        v8f acc[NT];
#pragma unroll
        for (int t = 0; t < NT; ++t)
            acc[t] = (v8f){0.f, 0.f, 0.f, 0.f, 0.f, 0.f, 0.f, 0.f};

        // ---- K loop: one A frag from LDS feeds 16 WMMAs per step ----
        for (int k = 0; k < DD; k += 32) {
            // A fragment (16-bit A layout: lane -> M=lo; hi selects K sub-chunks)
            const __bf16* ap = xs + lo * PADD + k + hi * 16;
            v8bf a0 = *(const v8bf*)(ap);
            v8bf a1 = *(const v8bf*)(ap + 8);
            v16bf a = __builtin_shufflevector(a0, a1, 0,1,2,3,4,5,6,7,8,9,10,11,12,13,14,15);
#pragma unroll
            for (int t = 0; t < NT; ++t) {
                const int j = colbase + t * 16 + lo;          // N = W1 row index
                const __bf16* bp = Wb + (size_t)j * DD + k + hi * 16;
                v8bf b0 = *(const v8bf*)(bp);
                v8bf b1 = *(const v8bf*)(bp + 8);
                v16bf b = __builtin_shufflevector(b0, b1, 0,1,2,3,4,5,6,7,8,9,10,11,12,13,14,15);
                acc[t] = __builtin_amdgcn_wmma_f32_16x16x32_bf16(
                    false, a, false, b, (short)0, acc[t], false, false);
            }
        }

        // ---- add b1, per-lane partial row statistics (rows M = r + 8*hi) ----
        float psum[8], psq[8];
#pragma unroll
        for (int r = 0; r < 8; ++r) { psum[r] = 0.f; psq[r] = 0.f; }
#pragma unroll
        for (int t = 0; t < NT; ++t) {
            float b1v = p.b1[br][colbase + t * 16 + lo];
#pragma unroll
            for (int r = 0; r < 8; ++r) {
                float h = acc[t][r] + b1v;
                acc[t][r] = h;
                psum[r] += h;
                psq[r]  += h * h;
            }
        }
        {
            float* slot = red + (wave * 32 + lane) * 16;
#pragma unroll
            for (int r = 0; r < 8; ++r) { slot[r] = psum[r]; slot[8 + r] = psq[r]; }
        }
        __syncthreads();

        // ---- reduce stats: thread m (<16) owns row m ----
        if (tid < ROWS) {
            int m = tid, h2 = m >> 3, r = m & 7;
            float s = 0.f, q = 0.f;
            for (int w = 0; w < 8; ++w)
                for (int l = 0; l < 16; ++l) {
                    const float* slot = red + ((w * 32 + h2 * 16 + l) * 16);
                    s += slot[r];
                    q += slot[8 + r];
                }
            float m_ = s * (1.f / DD);
            float v_ = q * (1.f / DD) - m_ * m_;
            mu[m] = m_;
            rs[m] = rsqrtf(v_ + EPSF);
        }
        __syncthreads();

        // ---- LN + ReLU + dot with W2 rows (fp32, on register-resident h) ----
        float muv[8], rsv[8];
#pragma unroll
        for (int r = 0; r < 8; ++r) { muv[r] = mu[r + 8 * hi]; rsv[r] = rs[r + 8 * hi]; }

        float d0[8], d1[8];
#pragma unroll
        for (int r = 0; r < 8; ++r) { d0[r] = 0.f; d1[r] = 0.f; }
#pragma unroll
        for (int t = 0; t < NT; ++t) {
            const int j = colbase + t * 16 + lo;
            float gv  = p.g[br][j];
            float bev = p.be[br][j];
            float w0  = p.W2[br][j];
            float w1  = p.W2[br][DD + j];
#pragma unroll
            for (int r = 0; r < 8; ++r) {
                float sc = rsv[r] * gv;
                float h  = fmaf(acc[t][r] - muv[r], sc, bev);
                h = fmaxf(h, 0.f);
                d0[r] = fmaf(h, w0, d0[r]);
                d1[r] = fmaf(h, w1, d1[r]);
            }
        }
        {
            float* slot = red + (wave * 32 + lane) * 16;
#pragma unroll
            for (int r = 0; r < 8; ++r) { slot[r] = d0[r]; slot[8 + r] = d1[r]; }
        }
        __syncthreads();

        if (tid < 32) {
            int m = tid >> 1, c = tid & 1;
            int h2 = m >> 3, r = m & 7;
            float s = p.b2[br][c];
            for (int w = 0; w < 8; ++w)
                for (int l = 0; l < 16; ++l)
                    s += red[(w * 32 + h2 * 16 + l) * 16 + c * 8 + r];
            logits[br][m][c] = s;
        }
        __syncthreads();
    }

    // ---- route on first-branch argmax (ties -> index 0), select fake/real ----
    if (tid < 32) {
        int m = tid >> 1, c = tid & 1;
        bool fake = logits[0][m][0] >= logits[0][m][1];   // argmax == 0
        float v = fake ? logits[1][m][c] : logits[2][m][c];
        p.out[(row0 + m) * 2 + c] = v;
    }
}

extern "C" void kernel_launch(void* const* d_in, const int* in_sizes, int n_in,
                              void* d_out, int out_size, void* d_ws, size_t ws_size,
                              hipStream_t stream) {
    // input order: x, then per branch {W1, b1, ln_g, ln_b, W2, b2} for first/fake/real
    const int W1E = DD * DD;                 // 4,194,304 elements
    __bf16* wsbf = (__bf16*)d_ws;            // needs 3 * W1E * 2 = 24 MB of workspace

    for (int b = 0; b < 3; ++b) {
        const float* w1 = (const float*)d_in[1 + 6 * b];
        cvt_f32_bf16<<<(W1E / 4 + 255) / 256, 256, 0, stream>>>(
            w1, wsbf + (size_t)b * W1E, W1E / 4);
    }

    Params p;
    p.x = (const float*)d_in[0];
    for (int b = 0; b < 3; ++b) {
        p.W1[b] = wsbf + (size_t)b * W1E;
        p.b1[b] = (const float*)d_in[2 + 6 * b];
        p.g[b]  = (const float*)d_in[3 + 6 * b];
        p.be[b] = (const float*)d_in[4 + 6 * b];
        p.W2[b] = (const float*)d_in[5 + 6 * b];
        p.b2[b] = (const float*)d_in[6 + 6 * b];
    }
    p.out = (float*)d_out;

    fused_moe_head<<<32768 / ROWS, 256, 0, stream>>>(p);
}
